// Block_42013370090004
// MI455X (gfx1250) — matrix operand order repaired and checked by reading, not compiled, not run on hardware
//
#include <hip/hip_runtime.h>
#include <hip/hip_bf16.h>
#include <math.h>

// ---- problem constants --------------------------------------------------
constexpr int B_    = 64;
constexpr int N_    = 229;   // 1 + O + P
constexpr int C_    = 1024;
constexpr int H_    = 16;
constexpr int D_    = 64;
constexpr int O_    = 32;
constexpr int HID_  = 4096;
constexpr int NT_   = 15;    // row tiles per batch: 2 prompt tiles + 13 pc tiles
constexpr int NPAD_ = 256;   // padded key count for attention / vT
constexpr float SCALE_ = 0.125f;  // D^-0.5

// ---- WMMA vector types ---------------------------------------------------
typedef __attribute__((ext_vector_type(16))) __bf16 bf16x16;
typedef __attribute__((ext_vector_type(8)))  __bf16 bf16x8;
typedef __attribute__((ext_vector_type(8)))  float  f32x8;

__device__ __forceinline__ bf16x8 ld8(const __bf16* p) {
  return *reinterpret_cast<const bf16x8*>(p);
}

__device__ __forceinline__ bf16x16 pack_a(const bf16x8& alo, const bf16x8& ahi) {
  bf16x16 a;
#pragma unroll
  for (int i = 0; i < 8; ++i) { a[i] = alo[i]; a[8 + i] = ahi[i]; }
  return a;
}

__device__ __forceinline__ f32x8 wmma_bf16(const bf16x16& a, const bf16x16& b,
                                           const f32x8& c) {
  return __builtin_amdgcn_wmma_f32_16x16x32_bf16(false, a, false, b,
                                                 (short)0, c, false, false);
}

// 16x64 output strip per wave: one A fragment reused across 4 B tiles.
// A: row-major bf16, one token row per lane (lane&15), K consumed 32/step.
// B: weight stored TRANSPOSED [Nout][K]; lane (= output col) reads 16
//    contiguous K values; tile j is 16 columns (16*ldk elements) apart.
template <int KSTEPS>
__device__ __forceinline__ void gemm4(const __bf16* __restrict__ Arow,
                                      const __bf16* __restrict__ Bcol0,
                                      size_t ldk, int lane, f32x8 acc[4]) {
  const int k0a = (lane < 16) ? 0 : 16;
  const int k0b = (lane < 16) ? 0 : 16;
#pragma unroll 2
  for (int s = 0; s < KSTEPS; ++s) {
    const int kk = s * 32;
    bf16x16 a = pack_a(ld8(Arow + kk + k0a), ld8(Arow + kk + 8 + k0a));
#pragma unroll
    for (int j = 0; j < 4; ++j) {
      bf16x16 bfrag = *reinterpret_cast<const bf16x16*>(
          Bcol0 + (size_t)j * 16 * ldk + kk + k0b);
      acc[j] = wmma_bf16(a, bfrag, acc[j]);
    }
  }
}

// Map (tile, row-in-tile) -> global token index.
// tiles 0..1: prompts (tokens 1..32). tiles 2..14: cls+patches
// slot 0 -> token 0, slot s in [1,196] -> token 32+s. slots >=197 invalid.
__device__ __forceinline__ void row_token(int tile, int m, int& tok, bool& valid) {
  if (tile < 2) { tok = 1 + tile * 16 + m; valid = true; }
  else {
    int s = (tile - 2) * 16 + m;
    valid = (s < 197);
    int sc = valid ? s : 196;
    tok = (sc == 0) ? 0 : (32 + sc);
  }
}

// ---- weight cast + transpose: w[K][Nc] f32 -> wt[Nc][K] bf16 -------------
__global__ void k_wcast(const float* __restrict__ w, __bf16* __restrict__ wt,
                        int K, int Nc) {
  size_t idx = (size_t)blockIdx.x * 256 + threadIdx.x;
  if (idx >= (size_t)K * Nc) return;
  int k = (int)(idx / Nc), n = (int)(idx % Nc);
  wt[(size_t)n * K + k] = (__bf16)w[idx];
}

__global__ void k_zero_bf16(__bf16* __restrict__ p, size_t n) {
  size_t i = (size_t)blockIdx.x * 256 + threadIdx.x;
  if (i < n) p[i] = (__bf16)0.0f;
}

// ---- LayerNorm (region-split params) + cast to bf16 ----------------------
__global__ __launch_bounds__(256) void k_ln(const float* __restrict__ x,
                                            const float* __restrict__ g,
                                            const float* __restrict__ be,
                                            const float* __restrict__ gp,
                                            const float* __restrict__ bep,
                                            __bf16* __restrict__ out) {
  int tf = blockIdx.x;              // b*N + token
  int tok = tf % N_;
  bool prompt = (tok >= 1 && tok < 1 + O_);
  const float* xr = x + (size_t)tf * C_;
  __shared__ float red[256];
  int tid = threadIdx.x;
  float vals[4];
  float s = 0.f;
#pragma unroll
  for (int i = 0; i < 4; ++i) { vals[i] = xr[tid + 256 * i]; s += vals[i]; }
  red[tid] = s; __syncthreads();
  for (int o = 128; o > 0; o >>= 1) { if (tid < o) red[tid] += red[tid + o]; __syncthreads(); }
  float mean = red[0] * (1.0f / C_);
  __syncthreads();
  float v = 0.f;
#pragma unroll
  for (int i = 0; i < 4; ++i) { float d = vals[i] - mean; v += d * d; }
  red[tid] = v; __syncthreads();
  for (int o = 128; o > 0; o >>= 1) { if (tid < o) red[tid] += red[tid + o]; __syncthreads(); }
  float rstd = rsqrtf(red[0] * (1.0f / C_) + 1e-5f);
  const float* gg = prompt ? gp : g;
  const float* bb = prompt ? bep : be;
  __bf16* orow = out + (size_t)tf * C_;
#pragma unroll
  for (int i = 0; i < 4; ++i) {
    int c = tid + 256 * i;
    orow[c] = (__bf16)((vals[i] - mean) * rstd * gg[c] + bb[c]);
  }
}

// ---- QKV GEMM: xn[B,N,C] x wT[3C,C] (region) -> q/k [B,H,N,D], vT [B,H,D,NPAD]
__global__ __launch_bounds__(256) void k_qkv(const __bf16* __restrict__ xn,
                                             const __bf16* __restrict__ wT,
                                             const __bf16* __restrict__ wTp,
                                             const float* __restrict__ bias,
                                             const float* __restrict__ biasp,
                                             __bf16* __restrict__ qb,
                                             __bf16* __restrict__ kb,
                                             __bf16* __restrict__ vT) {
  const int NCH = (3 * C_) / 512;   // 6
  int bid = blockIdx.x;
  int chunk = bid % NCH; int t = bid / NCH;
  int tile = t % NT_;   int b = t / NT_;
  int wave = threadIdx.x >> 5, lane = threadIdx.x & 31;
  int n = lane & 15;
  int colbase = chunk * 512 + wave * 64;
  bool prompt = tile < 2;
  int atok; bool avalid;
  row_token(tile, n, atok, avalid);
  const __bf16* Arow = xn + ((size_t)b * N_ + atok) * C_;
  const __bf16* Bcol0 = (prompt ? wTp : wT) + (size_t)(colbase + n) * C_;
  f32x8 acc[4] = {{0.f}, {0.f}, {0.f}, {0.f}};
  gemm4<C_ / 32>(Arow, Bcol0, C_, lane, acc);
  const float* bsel = prompt ? biasp : bias;
#pragma unroll
  for (int j = 0; j < 4; ++j) {
    int col = colbase + j * 16 + n;
    float bi = bsel[col];
    int which = col >> 10;       // 0=q 1=k 2=v
    int cc = col & 1023;
    int hh = cc >> 6, d = cc & 63;
    size_t bh = (size_t)b * H_ + hh;
#pragma unroll
    for (int r = 0; r < 8; ++r) {
      int mrow = r + ((lane < 16) ? 0 : 8);
      int tok; bool valid;
      row_token(tile, mrow, tok, valid);
      if (!valid) continue;
      float v = acc[j][r] + bi;
      if (which == 0)      qb[(bh * N_ + tok) * D_ + d] = (__bf16)v;
      else if (which == 1) kb[(bh * N_ + tok) * D_ + d] = (__bf16)v;
      else                 vT[(bh * D_ + d) * NPAD_ + tok] = (__bf16)v;
    }
  }
}

// ---- attention: one WG per (b, h, q-tile) --------------------------------
__global__ __launch_bounds__(256) void k_attn(const __bf16* __restrict__ qb,
                                              const __bf16* __restrict__ kb,
                                              const __bf16* __restrict__ vT,
                                              __bf16* __restrict__ ob) {
  __shared__ float  S[16][NPAD_];
  __shared__ __bf16 Pm[16][NPAD_];
  int bid = blockIdx.x;
  int qt = bid % NT_; int t = bid / NT_;
  int hh = t % H_;    int b = t / H_;
  int tid = threadIdx.x, wave = tid >> 5, lane = tid & 31;
  size_t bh = (size_t)b * H_ + hh;

  float* Sf = &S[0][0];
  for (int i = tid; i < 16 * NPAD_; i += 256) Sf[i] = -__builtin_inff();
  __syncthreads();

  // phase 1: S = Q K^T * scale (+mask); Q fragments hoisted out of key loop
  {
    int m = lane & 15;
    int k0a = (lane < 16) ? 0 : 16;
    int k0b = (lane < 16) ? 0 : 16;
    int qtok = qt * 16 + m;
    int qc = qtok < N_ ? qtok : N_ - 1;
    const __bf16* Arow = qb + (bh * N_ + qc) * D_;
    bf16x16 qa0 = pack_a(ld8(Arow + k0a),      ld8(Arow + 8 + k0a));
    bf16x16 qa1 = pack_a(ld8(Arow + 32 + k0a), ld8(Arow + 40 + k0a));
    for (int kt = wave; kt < NT_; kt += 8) {
      int key = kt * 16 + m;
      int kc = key < N_ ? key : N_ - 1;
      const __bf16* Brow = kb + (bh * N_ + kc) * D_;
      bf16x16 b0 = *reinterpret_cast<const bf16x16*>(Brow + k0b);
      bf16x16 b1 = *reinterpret_cast<const bf16x16*>(Brow + 32 + k0b);
      f32x8 acc = {0.f, 0.f, 0.f, 0.f, 0.f, 0.f, 0.f, 0.f};
      acc = wmma_bf16(qa0, b0, acc);
      acc = wmma_bf16(qa1, b1, acc);
      bool kvalid = key < N_;
      bool kprompt = (key >= 1 && key < 1 + O_);
#pragma unroll
      for (int r = 0; r < 8; ++r) {
        int mrow = r + ((lane < 16) ? 0 : 8);
        int qrow = qt * 16 + mrow;
        bool qprompt = (qrow >= 1 && qrow < 1 + O_);
        float v = acc[r] * SCALE_;
        if (!kvalid || (kprompt && !qprompt)) v = -__builtin_inff();
        S[mrow][key] = v;
      }
    }
  }
  __syncthreads();

  // phase 2: row softmax -> Pm (bf16)
  {
    int row = tid >> 4, c0 = tid & 15;
    float mx = -__builtin_inff();
#pragma unroll
    for (int j = 0; j < 16; ++j) mx = fmaxf(mx, S[row][c0 + 16 * j]);
    for (int o = 8; o; o >>= 1) mx = fmaxf(mx, __shfl_xor(mx, o, 32));
    float pv[16];
    float sum = 0.f;
#pragma unroll
    for (int j = 0; j < 16; ++j) { pv[j] = expf(S[row][c0 + 16 * j] - mx); sum += pv[j]; }
    for (int o = 8; o; o >>= 1) sum += __shfl_xor(sum, o, 32);
    float inv = 1.0f / sum;
#pragma unroll
    for (int j = 0; j < 16; ++j) Pm[row][c0 + 16 * j] = (__bf16)(pv[j] * inv);
  }
  __syncthreads();

  // phase 3: O = P V   (waves 0..3, one 16-wide d-tile each)
  if (wave < 4) {
    int dt = wave;
    int m = lane & 15;
    int k0a = (lane < 16) ? 0 : 16;
    int k0b = (lane < 16) ? 0 : 16;
    int d = dt * 16 + (lane & 15);
    const __bf16* Bcol = vT + (bh * D_ + d) * NPAD_;
    f32x8 acc = {0.f, 0.f, 0.f, 0.f, 0.f, 0.f, 0.f, 0.f};
#pragma unroll 2
    for (int s = 0; s < NPAD_ / 32; ++s) {
      int kk = s * 32;
      bf16x8 alo, ahi;
#pragma unroll
      for (int i = 0; i < 8; ++i) {
        alo[i] = Pm[m][kk + k0a + i];
        ahi[i] = Pm[m][kk + 8 + k0a + i];
      }
      bf16x16 bfrag = *reinterpret_cast<const bf16x16*>(Bcol + kk + k0b);
      acc = wmma_bf16(pack_a(alo, ahi), bfrag, acc);
    }
#pragma unroll
    for (int r = 0; r < 8; ++r) {
      int mrow = r + ((lane < 16) ? 0 : 8);
      int qtok = qt * 16 + mrow;
      if (qtok < N_) ob[((size_t)b * N_ + qtok) * C_ + hh * D_ + d] = (__bf16)acc[r];
    }
  }
}

// ---- proj GEMM: ob x wT_proj (region) + bias + x -> x1 (f32) -------------
__global__ __launch_bounds__(256) void k_proj(const __bf16* __restrict__ ob,
                                              const __bf16* __restrict__ wT,
                                              const __bf16* __restrict__ wTp,
                                              const float* __restrict__ bias,
                                              const float* __restrict__ biasp,
                                              const float* __restrict__ x,
                                              float* __restrict__ x1) {
  const int NCH = C_ / 512;  // 2
  int bid = blockIdx.x;
  int chunk = bid % NCH; int t = bid / NCH;
  int tile = t % NT_;    int b = t / NT_;
  int wave = threadIdx.x >> 5, lane = threadIdx.x & 31;
  int n = lane & 15;
  int colbase = chunk * 512 + wave * 64;
  bool prompt = tile < 2;
  int atok; bool avalid;
  row_token(tile, n, atok, avalid);
  const __bf16* Arow = ob + ((size_t)b * N_ + atok) * C_;
  const __bf16* Bcol0 = (prompt ? wTp : wT) + (size_t)(colbase + n) * C_;
  f32x8 acc[4] = {{0.f}, {0.f}, {0.f}, {0.f}};
  gemm4<C_ / 32>(Arow, Bcol0, C_, lane, acc);
  const float* bsel = prompt ? biasp : bias;
#pragma unroll
  for (int j = 0; j < 4; ++j) {
    int col = colbase + j * 16 + n;
    float bi = bsel[col];
#pragma unroll
    for (int r = 0; r < 8; ++r) {
      int mrow = r + ((lane < 16) ? 0 : 8);
      int tok; bool valid;
      row_token(tile, mrow, tok, valid);
      if (!valid) continue;
      size_t off = ((size_t)b * N_ + tok) * C_ + col;
      x1[off] = x[off] + acc[j][r] + bi;
    }
  }
}

// ---- FC1 GEMM + exact GELU: x1n x fc1T (region) -> h bf16 [B,N,HID] ------
__global__ __launch_bounds__(256) void k_fc1(const __bf16* __restrict__ x1n,
                                             const __bf16* __restrict__ wT,
                                             const __bf16* __restrict__ wTp,
                                             const float* __restrict__ bias,
                                             const float* __restrict__ biasp,
                                             __bf16* __restrict__ hbuf) {
  const int NCH = HID_ / 512;  // 8
  int bid = blockIdx.x;
  int chunk = bid % NCH; int t = bid / NCH;
  int tile = t % NT_;    int b = t / NT_;
  int wave = threadIdx.x >> 5, lane = threadIdx.x & 31;
  int n = lane & 15;
  int colbase = chunk * 512 + wave * 64;
  bool prompt = tile < 2;
  int atok; bool avalid;
  row_token(tile, n, atok, avalid);
  const __bf16* Arow = x1n + ((size_t)b * N_ + atok) * C_;
  const __bf16* Bcol0 = (prompt ? wTp : wT) + (size_t)(colbase + n) * C_;
  f32x8 acc[4] = {{0.f}, {0.f}, {0.f}, {0.f}};
  gemm4<C_ / 32>(Arow, Bcol0, C_, lane, acc);
  const float* bsel = prompt ? biasp : bias;
#pragma unroll
  for (int j = 0; j < 4; ++j) {
    int col = colbase + j * 16 + n;
    float bi = bsel[col];
#pragma unroll
    for (int r = 0; r < 8; ++r) {
      int mrow = r + ((lane < 16) ? 0 : 8);
      int tok; bool valid;
      row_token(tile, mrow, tok, valid);
      if (!valid) continue;
      float v = acc[j][r] + bi;
      float g = 0.5f * v * (1.0f + erff(v * 0.70710678118654752f));
      hbuf[((size_t)b * N_ + tok) * HID_ + col] = (__bf16)g;
    }
  }
}

// ---- FC2 GEMM: h x fc2T (region) + bias + x1 -> out (f32) ----------------
__global__ __launch_bounds__(256) void k_fc2(const __bf16* __restrict__ hbuf,
                                             const __bf16* __restrict__ wT,
                                             const __bf16* __restrict__ wTp,
                                             const float* __restrict__ bias,
                                             const float* __restrict__ biasp,
                                             const float* __restrict__ x1,
                                             float* __restrict__ out) {
  const int NCH = C_ / 512;  // 2
  int bid = blockIdx.x;
  int chunk = bid % NCH; int t = bid / NCH;
  int tile = t % NT_;    int b = t / NT_;
  int wave = threadIdx.x >> 5, lane = threadIdx.x & 31;
  int n = lane & 15;
  int colbase = chunk * 512 + wave * 64;
  bool prompt = tile < 2;
  int atok; bool avalid;
  row_token(tile, n, atok, avalid);
  const __bf16* Arow = hbuf + ((size_t)b * N_ + atok) * HID_;
  const __bf16* Bcol0 = (prompt ? wTp : wT) + (size_t)(colbase + n) * HID_;
  f32x8 acc[4] = {{0.f}, {0.f}, {0.f}, {0.f}};
  gemm4<HID_ / 32>(Arow, Bcol0, HID_, lane, acc);
  const float* bsel = prompt ? biasp : bias;
#pragma unroll
  for (int j = 0; j < 4; ++j) {
    int col = colbase + j * 16 + n;
    float bi = bsel[col];
#pragma unroll
    for (int r = 0; r < 8; ++r) {
      int mrow = r + ((lane < 16) ? 0 : 8);
      int tok; bool valid;
      row_token(tile, mrow, tok, valid);
      if (!valid) continue;
      size_t off = ((size_t)b * N_ + tok) * C_ + col;
      out[off] = x1[off] + acc[j][r] + bi;
    }
  }
}

// ---- host ----------------------------------------------------------------
extern "C" void kernel_launch(void* const* d_in, const int* in_sizes, int n_in,
                              void* d_out, int out_size, void* d_ws, size_t ws_size,
                              hipStream_t stream) {
  const float* x        = (const float*)d_in[0];
  const float* w_qkv    = (const float*)d_in[1];
  const float* b_qkv    = (const float*)d_in[2];
  const float* w_proj   = (const float*)d_in[3];
  const float* b_proj   = (const float*)d_in[4];
  const float* w_qkv_p  = (const float*)d_in[5];
  const float* b_qkv_p  = (const float*)d_in[6];
  const float* w_proj_p = (const float*)d_in[7];
  const float* b_proj_p = (const float*)d_in[8];
  const float* n1_g  = (const float*)d_in[9];
  const float* n1_b  = (const float*)d_in[10];
  const float* n1p_g = (const float*)d_in[11];
  const float* n1p_b = (const float*)d_in[12];
  const float* n2_g  = (const float*)d_in[13];
  const float* n2_b  = (const float*)d_in[14];
  const float* n2p_g = (const float*)d_in[15];
  const float* n2p_b = (const float*)d_in[16];
  const float* fc1_w  = (const float*)d_in[17];
  const float* fc1_b  = (const float*)d_in[18];
  const float* fc2_w  = (const float*)d_in[19];
  const float* fc2_b  = (const float*)d_in[20];
  const float* pfc1_w = (const float*)d_in[21];
  const float* pfc1_b = (const float*)d_in[22];
  const float* pfc2_w = (const float*)d_in[23];
  const float* pfc2_b = (const float*)d_in[24];
  float* outp = (float*)d_out;

  char* ws = (char*)d_ws;
  size_t off = 0;
  auto alloc = [&](size_t bytes) -> void* {
    void* p = ws + off;
    off = (off + bytes + 255) & ~(size_t)255;
    return p;
  };

  const size_t BNC = (size_t)B_ * N_ * C_;
  __bf16* wT_qkv   = (__bf16*)alloc((size_t)3 * C_ * C_ * 2);
  __bf16* wT_qkv_p = (__bf16*)alloc((size_t)3 * C_ * C_ * 2);
  __bf16* wT_proj  = (__bf16*)alloc((size_t)C_ * C_ * 2);
  __bf16* wT_proj_p= (__bf16*)alloc((size_t)C_ * C_ * 2);
  __bf16* wT_fc1   = (__bf16*)alloc((size_t)HID_ * C_ * 2);
  __bf16* wT_pfc1  = (__bf16*)alloc((size_t)HID_ * C_ * 2);
  __bf16* wT_fc2   = (__bf16*)alloc((size_t)C_ * HID_ * 2);
  __bf16* wT_pfc2  = (__bf16*)alloc((size_t)C_ * HID_ * 2);
  __bf16* xn   = (__bf16*)alloc(BNC * 2);
  __bf16* qb   = (__bf16*)alloc(BNC * 2);
  __bf16* kb   = (__bf16*)alloc(BNC * 2);
  __bf16* vT   = (__bf16*)alloc((size_t)B_ * H_ * D_ * NPAD_ * 2);
  __bf16* ob   = (__bf16*)alloc(BNC * 2);
  float*  x1   = (float*)alloc(BNC * 4);
  __bf16* x1n  = (__bf16*)alloc(BNC * 2);
  __bf16* hbuf = (__bf16*)alloc((size_t)B_ * N_ * HID_ * 2);

  auto cast_w = [&](const float* w, __bf16* wt, int K, int Nc) {
    size_t n = (size_t)K * Nc;
    k_wcast<<<(unsigned)((n + 255) / 256), 256, 0, stream>>>(w, wt, K, Nc);
  };
  cast_w(w_qkv,    wT_qkv,   C_,   3 * C_);
  cast_w(w_qkv_p,  wT_qkv_p, C_,   3 * C_);
  cast_w(w_proj,   wT_proj,  C_,   C_);
  cast_w(w_proj_p, wT_proj_p,C_,   C_);
  cast_w(fc1_w,    wT_fc1,   C_,   HID_);
  cast_w(pfc1_w,   wT_pfc1,  C_,   HID_);
  cast_w(fc2_w,    wT_fc2,   HID_, C_);
  cast_w(pfc2_w,   wT_pfc2,  HID_, C_);

  size_t vtn = (size_t)B_ * H_ * D_ * NPAD_;
  k_zero_bf16<<<(unsigned)((vtn + 255) / 256), 256, 0, stream>>>(vT, vtn);

  k_ln<<<B_ * N_, 256, 0, stream>>>(x, n1_g, n1_b, n1p_g, n1p_b, xn);

  k_qkv<<<B_ * NT_ * ((3 * C_) / 512), 256, 0, stream>>>(
      xn, wT_qkv, wT_qkv_p, b_qkv, b_qkv_p, qb, kb, vT);

  k_attn<<<B_ * H_ * NT_, 256, 0, stream>>>(qb, kb, vT, ob);

  k_proj<<<B_ * NT_ * (C_ / 512), 256, 0, stream>>>(
      ob, wT_proj, wT_proj_p, b_proj, b_proj_p, x, x1);

  k_ln<<<B_ * N_, 256, 0, stream>>>(x1, n2_g, n2_b, n2p_g, n2p_b, x1n);

  k_fc1<<<B_ * NT_ * (HID_ / 512), 256, 0, stream>>>(
      x1n, wT_fc1, wT_pfc1, fc1_b, pfc1_b, hbuf);

  k_fc2<<<B_ * NT_ * (C_ / 512), 256, 0, stream>>>(
      hbuf, wT_fc2, wT_pfc2, fc2_b, pfc2_b, x1, outp);
}
